// roleReLU_42296837931316
// MI455X (gfx1250) — matrix operand order repaired and checked by reading, not compiled, not run on hardware
//
#include <hip/hip_runtime.h>
#include <hip/hip_bf16.h>

// ---- problem constants (from reference) ----
constexpr int kS   = 160;   // sequence length
constexpr int kH   = 768;   // hidden
constexpr int kO   = 260;   // REL_SIZE * TAG_SIZE
constexpr int kOT  = 17;    // o tiles of 16 (260 padded to 272)
constexpr int kKS  = 24;    // k-steps: 768 / 32

typedef _Float16 v16h __attribute__((ext_vector_type(16)));
typedef _Float16 v8h  __attribute__((ext_vector_type(8)));
typedef float    v8f  __attribute__((ext_vector_type(8)));

static __device__ __forceinline__ v16h pack16(v8h lo, v8h hi) {
    v16h r;
#pragma unroll
    for (int e = 0; e < 8; ++e) { r[e] = lo[e]; r[8 + e] = hi[e]; }
    return r;
}

// ------------------------------------------------------------------
// Kernel 0: RoPE -> f16.  yrot[s,2p]   = y[s,2p]*cos - y[s,2p+1]*sin
//                         yrot[s,2p+1] = y[s,2p+1]*cos + y[s,2p]*sin
// theta_p(s) = s * 10000^(-p/384)
// ------------------------------------------------------------------
__global__ void rope_k(const float* __restrict__ y, _Float16* __restrict__ yrot) {
    int t = blockIdx.x * blockDim.x + threadIdx.x;        // over S * H/2
    if (t >= kS * (kH / 2)) return;
    int s = t / (kH / 2);
    int p = t - s * (kH / 2);
    float inv   = __expf(-(float)p * (9.210340371976184f / 384.0f));
    float theta = (float)s * inv;
    float c = __cosf(theta), sn = __sinf(theta);
    float e0 = y[s * kH + 2 * p];
    float e1 = y[s * kH + 2 * p + 1];
    yrot[s * kH + 2 * p]     = (_Float16)(e0 * c - e1 * sn);
    yrot[s * kH + 2 * p + 1] = (_Float16)(e1 * c + e0 * sn);
}

// ------------------------------------------------------------------
// Kernel C1: W1 fp32 -> f16, layout preserved ([o][1536], k-contiguous)
// ------------------------------------------------------------------
__global__ void w1half_k(const float* __restrict__ W1, _Float16* __restrict__ w1h) {
    int t = blockIdx.x * blockDim.x + threadIdx.x;
    if (t < kH * 2 * kH) w1h[t] = (_Float16)W1[t];
}

// ------------------------------------------------------------------
// Kernel 1: WMMA GEMM for left/right.
// Virtual output column c in [0,1536): c<768 -> left[s,c] (B col = W1[c, k]),
// c>=768 -> right[s,c-768] + b1 (B col = W1[c-768, 768+k]).
// Grid (10 s-tiles, 12), 8 waves/block, wave owns o-tile blockIdx.y*8+wave.
// B fragments read directly from w1h: 16 contiguous halves per lane.
// ------------------------------------------------------------------
__global__ void __launch_bounds__(256)
lr_gemm_k(const _Float16* __restrict__ yrot, const _Float16* __restrict__ w1h,
          const float* __restrict__ b1, float* __restrict__ left,
          float* __restrict__ right) {
    const int tid  = threadIdx.x;
    const int wave = tid >> 5;
    const int lane = tid & 31;
    const int m    = lane & 15;      // A row (s_local) == D column (c_local)
    const int hi   = lane >> 4;

    const int s0 = blockIdx.x * 16;
    const int ct = blockIdx.y * 8 + wave;     // 0..95 column tiles of 16
    const int c  = ct * 16 + m;               // this lane's output column

    // B column base in w1h (row of W1, k-contiguous)
    const _Float16* bcol = (c < kH) ? (w1h + (size_t)c * (2 * kH))
                                    : (w1h + (size_t)(c - kH) * (2 * kH) + kH);
    const _Float16* arow = yrot + (size_t)(s0 + m) * kH + hi * 16;

    v8f acc = {};
#pragma unroll
    for (int ks = 0; ks < kKS; ++ks) {
        // A fragment: row m, k = ks*32 + 16*hi + {0..15} (contiguous)
        v8h alo = *(const v8h*)(arow + ks * 32);
        v8h ahi = *(const v8h*)(arow + ks * 32 + 8);
        // B fragment: column c, k = ks*32 + 16*hi + {0..15} (contiguous)
        const _Float16* bp = bcol + ks * 32 + hi * 16;
        v8h blo = *(const v8h*)(bp);
        v8h bhi = *(const v8h*)(bp + 8);
        acc = __builtin_amdgcn_wmma_f32_16x16x32_f16(
                  false, pack16(alo, ahi), false, pack16(blo, bhi),
                  (short)0, acc, false, false);
    }
    // D: row s_local = e + 8*hi, col = m  (c computed above)
#pragma unroll
    for (int e = 0; e < 8; ++e) {
        int s = s0 + e + 8 * hi;
        if (c < kH) left[(size_t)s * kH + c] = acc[e];
        else        right[(size_t)s * kH + (c - kH)] = acc[e] + b1[c - kH];
    }
}

// ------------------------------------------------------------------
// Kernel W: pre-swizzle W2 (260x768 fp32) into f16 WMMA B-fragments,
// zero-padded to 272 columns.  16 contiguous halves per (o_tile,ks,lane).
// ------------------------------------------------------------------
__global__ void w2frag_k(const float* __restrict__ W2, _Float16* __restrict__ frag) {
    int t = blockIdx.x * blockDim.x + threadIdx.x;
    const int TOT = kOT * kKS * 32 * 16;
    if (t >= TOT) return;
    int e    = t & 15;
    int r    = t >> 4;
    int lane = r & 31;
    int r2   = r >> 5;
    int ks   = r2 % kKS;
    int ot   = r2 / kKS;
    int n  = lane & 15;
    int kk = ks * 32 + ((lane >> 4) * 16) + e;
    int o  = ot * 16 + n;
    float v = (o < kO) ? W2[(size_t)o * kH + kk] : 0.0f;
    frag[t] = (_Float16)v;
}

// ------------------------------------------------------------------
// Kernel 2: pair-GEMM.  Block = (i, j_tile).  A[j,k] = relu(left[i,k] +
// right[j0+j,k]) built once in LDS (16x768 f16 = 24KB).  K-outer loop:
// each wave owns o-tiles (w, w+8) with two accumulators sharing one A
// fragment per k-step; B fragments software-pipelined one step ahead.
// Tiles 0..15 cover o<256 (no bounds check); tile 16 is a short epilogue
// pass on wave 0.  Epilogue: +b2, sigmoid, scatter to (65,160,160,4).
// ------------------------------------------------------------------
__device__ __forceinline__ void store_tile(v8f c, int ot, int m, int hi,
                                           int i, int j0,
                                           const float* __restrict__ b2,
                                           float* __restrict__ out) {
#pragma unroll
    for (int e = 0; e < 8; ++e) {
        int o = ot * 16 + m;
        if (o < kO) {
            float x  = c[e] + b2[o];
            float sg = 1.0f / (1.0f + __expf(-x));
            int rr = o >> 2, tt = o & 3;
            int j  = j0 + e + 8 * hi;
            out[(((size_t)rr * kS + i) * kS + j) * 4 + tt] = sg;
        }
    }
}

__global__ void __launch_bounds__(256)
pair_gemm_k(const float* __restrict__ left, const float* __restrict__ right,
            const _Float16* __restrict__ w2f, const float* __restrict__ b2,
            float* __restrict__ out) {
    __shared__ alignas(16) _Float16 Alds[16 * kH];   // 24 KB
    const int i   = blockIdx.x;
    const int j0  = blockIdx.y * 16;
    const int tid = threadIdx.x;

    // Cooperative A-tile build (consecutive tids -> consecutive k).
    for (int idx = tid; idx < 16 * kH; idx += 256) {
        int j = idx / kH;
        int k = idx - j * kH;
        float v = left[i * kH + k] + right[(j0 + j) * kH + k];
        Alds[idx] = (_Float16)(v > 0.f ? v : 0.f);
    }
    __syncthreads();

    const int wave = tid >> 5;
    const int lane = tid & 31;
    const int m    = lane & 15;
    const int hi   = lane >> 4;

    // per-(tile,lane) fragment streams; stride per k-step = 32*16 halves
    const _Float16* b0base = w2f + ((size_t)wave * kKS * 32 + lane) * 16;
    const _Float16* b1base = w2f + ((size_t)(wave + 8) * kKS * 32 + lane) * 16;

    v8f c0 = {}, c1 = {};
    v8h b0lo = *(const v8h*)(b0base),      b0hi = *(const v8h*)(b0base + 8);
    v8h b1lo = *(const v8h*)(b1base),      b1hi = *(const v8h*)(b1base + 8);

#pragma unroll
    for (int ks = 0; ks < kKS; ++ks) {
        // prefetch next step's B fragments before consuming current ones
        v8h n0lo = b0lo, n0hi = b0hi, n1lo = b1lo, n1hi = b1hi;
        if (ks < kKS - 1) {
            const _Float16* p0 = b0base + (size_t)(ks + 1) * 512;
            const _Float16* p1 = b1base + (size_t)(ks + 1) * 512;
            n0lo = *(const v8h*)(p0);  n0hi = *(const v8h*)(p0 + 8);
            n1lo = *(const v8h*)(p1);  n1hi = *(const v8h*)(p1 + 8);
        }
        const _Float16* ap = &Alds[m * kH + ks * 32 + hi * 16];
        v16h a = pack16(*(const v8h*)(ap), *(const v8h*)(ap + 8));

        c0 = __builtin_amdgcn_wmma_f32_16x16x32_f16(
                 false, a, false, pack16(b0lo, b0hi), (short)0, c0, false, false);
        c1 = __builtin_amdgcn_wmma_f32_16x16x32_f16(
                 false, a, false, pack16(b1lo, b1hi), (short)0, c1, false, false);

        b0lo = n0lo; b0hi = n0hi; b1lo = n1lo; b1hi = n1hi;
    }
    store_tile(c0, wave,     m, hi, i, j0, b2, out);
    store_tile(c1, wave + 8, m, hi, i, j0, b2, out);

    // leftover tile 16 (o = 256..271; only 256..259 stored) on wave 0
    if (wave == 0) {
        const _Float16* bt = w2f + ((size_t)16 * kKS * 32 + lane) * 16;
        v8f c2 = {};
#pragma unroll
        for (int ks = 0; ks < kKS; ++ks) {
            const _Float16* ap = &Alds[m * kH + ks * 32 + hi * 16];
            v16h a = pack16(*(const v8h*)(ap), *(const v8h*)(ap + 8));
            const _Float16* bp = bt + (size_t)ks * 512;
            c2 = __builtin_amdgcn_wmma_f32_16x16x32_f16(
                     false, a, false, pack16(*(const v8h*)(bp), *(const v8h*)(bp + 8)),
                     (short)0, c2, false, false);
        }
        store_tile(c2, 16, m, hi, i, j0, b2, out);
    }
}

// ------------------------------------------------------------------
extern "C" void kernel_launch(void* const* d_in, const int* in_sizes, int n_in,
                              void* d_out, int out_size, void* d_ws, size_t ws_size,
                              hipStream_t stream) {
    const float* y  = (const float*)d_in[0];
    // d_in[1] = event_idx (unused by the reference)
    const float* W1 = (const float*)d_in[2];
    const float* b1 = (const float*)d_in[3];
    const float* W2 = (const float*)d_in[4];
    const float* b2 = (const float*)d_in[5];
    float* out = (float*)d_out;

    char* ws = (char*)d_ws;
    _Float16* yroth = (_Float16*)(ws);                    // 160*768*2   =   245,760 B
    _Float16* w1h   = (_Float16*)(ws + 245760);           // 768*1536*2  = 2,359,296 B
    float*    left  = (float*)(ws + 2605056);             //               491,520 B
    float*    right = (float*)(ws + 3096576);             //               491,520 B
    _Float16* w2f   = (_Float16*)(ws + 3588096);          // 17*24*32*16*2 = 417,792 B

    rope_k<<<dim3((kS * (kH / 2) + 255) / 256), dim3(256), 0, stream>>>(y, yroth);
    w1half_k<<<dim3((kH * 2 * kH + 255) / 256), dim3(256), 0, stream>>>(W1, w1h);
    w2frag_k<<<dim3((kOT * kKS * 32 * 16 + 255) / 256), dim3(256), 0, stream>>>(W2, w2f);
    lr_gemm_k<<<dim3(10, 12), dim3(256), 0, stream>>>(yroth, w1h, b1, left, right);
    pair_gemm_k<<<dim3(kS, 10), dim3(256), 0, stream>>>(left, right, w2f, b2, out);
}
